// PointGiraffeLayer_64295660421508
// MI455X (gfx1250) — compile-verified, matrix operand order checked by reading, not run on hardware
//
#include <hip/hip_runtime.h>
#include <hip/hip_bf16.h>
#include <math.h>

// ---------------------------------------------------------------------------
// PointGiraffe FP layer for MI455X (gfx1250, wave32).
// - GEMMs on v_wmma_f32_16x16x32_f16 (f16 A/B, f32 accum).
// - Weights pre-swizzled once per launch into WMMA B-fragment order (f16),
//   so the GEMM B-load is two coalesced global_load_b128 with no VALU cvt.
// - Each wave owns a 16-row x 64-col strip: A fragment converted once per
//   K-step, reused by 4 WMMAs (4 accumulators).
// - BN stats fused into GEMM-1 epilogue (global f32 atomics); BN+ReLU fused
//   into GEMM-2's A-operand load.
// - kNN: VALU pass, LDS-staged 256-point source tiles + global_prefetch.
// ---------------------------------------------------------------------------

typedef __attribute__((ext_vector_type(16))) _Float16 v16h;
typedef __attribute__((ext_vector_type(8)))  float    v8f;

#define CDIM 128

// ---------------------------- 3-NN search ----------------------------------
__global__ void knn3_kernel(const float* __restrict__ tgt,
                            const float* __restrict__ src,
                            int Nt, int Ns,
                            int* __restrict__ oidx, float* __restrict__ ow) {
  __shared__ float sx[256], sy[256], sz[256];
  const int t  = blockIdx.x * 256 + threadIdx.x;
  const int b  = t / Nt;            // uniform across the block
  const float* tp = tgt + (size_t)t * 3;
  const float tx = tp[0], ty = tp[1], tz = tp[2];
  const float* sp = src + (size_t)b * Ns * 3;

  float d0 = 3.0e38f, d1 = 3.0e38f, d2 = 3.0e38f;
  int   i0 = 0, i1 = 0, i2 = 0;

  for (int base = 0; base < Ns; base += 256) {
    const int j = threadIdx.x;
    sx[j] = sp[(size_t)(base + j) * 3 + 0];
    sy[j] = sp[(size_t)(base + j) * 3 + 1];
    sz[j] = sp[(size_t)(base + j) * 3 + 2];
    if (base + 256 < Ns)            // gfx1250 global_prefetch_b8 of next tile
      __builtin_prefetch(sp + (size_t)(base + 256 + j) * 3, 0, 0);
    __syncthreads();
#pragma unroll 4
    for (int k = 0; k < 256; ++k) {
      const float dx = tx - sx[k];
      const float dy = ty - sy[k];
      const float dz = tz - sz[k];
      const float d  = dx * dx + dy * dy + dz * dz;
      const int  id  = base + k;
      if (d < d2) {                 // strict < keeps earliest index on ties
        if (d < d1) {
          d2 = d1; i2 = i1;
          if (d < d0) { d1 = d0; i1 = i0; d0 = d; i0 = id; }
          else        { d1 = d;  i1 = id; }
        } else { d2 = d; i2 = id; }
      }
    }
    __syncthreads();
  }
  float w0 = 1.0f / (sqrtf(fmaxf(d0, 0.0f)) + 1e-8f);
  float w1 = 1.0f / (sqrtf(fmaxf(d1, 0.0f)) + 1e-8f);
  float w2 = 1.0f / (sqrtf(fmaxf(d2, 0.0f)) + 1e-8f);
  const float inv = 1.0f / (w0 + w1 + w2);
  ow[(size_t)t * 3 + 0] = w0 * inv;
  ow[(size_t)t * 3 + 1] = w1 * inv;
  ow[(size_t)t * 3 + 2] = w2 * inv;
  oidx[(size_t)t * 3 + 0] = b * Ns + i0;   // global feature-row indices
  oidx[(size_t)t * 3 + 1] = b * Ns + i1;
  oidx[(size_t)t * 3 + 2] = b * Ns + i2;
}

// ------------------------- weighted 3-gather -------------------------------
__global__ void interp3_kernel(const float* __restrict__ feats,
                               const int* __restrict__ idx,
                               const float* __restrict__ w,
                               float* __restrict__ out, int ntTotal) {
  const int gid = blockIdx.x * blockDim.x + threadIdx.x;
  const int t = gid >> 5;
  if (t >= ntTotal) return;
  const int cg = gid & 31;
  const int j0 = idx[(size_t)t * 3 + 0];
  const int j1 = idx[(size_t)t * 3 + 1];
  const int j2 = idx[(size_t)t * 3 + 2];
  const float w0 = w[(size_t)t * 3 + 0];
  const float w1 = w[(size_t)t * 3 + 1];
  const float w2 = w[(size_t)t * 3 + 2];
  const float4* f = (const float4*)feats;
  const float4 a = f[(size_t)j0 * 32 + cg];
  const float4 b = f[(size_t)j1 * 32 + cg];
  const float4 c = f[(size_t)j2 * 32 + cg];
  float4 r;
  r.x = w0 * a.x + w1 * b.x + w2 * c.x;
  r.y = w0 * a.y + w1 * b.y + w2 * c.y;
  r.z = w0 * a.z + w1 * b.z + w2 * c.z;
  r.w = w0 * a.w + w1 * b.w + w2 * c.w;
  ((float4*)out)[(size_t)t * 32 + cg] = r;
}

// ------------------- weight pre-swizzle to WMMA B fragments ----------------
// W: (128 x K) row-major fp32.  Output fragment order:
//   frag[(nt*KS + ks)*32 + lane] = 16 halves for B column n = nt*16+(lane&15),
//   K-halves 0..7 -> k = kh..kh+7, 8..15 -> kh+16..kh+23, kh = ks*32+8*(lane>>4)
__global__ void packW_kernel(const float* __restrict__ W,
                             _Float16* __restrict__ out, int K) {
  const int tid  = blockIdx.x * blockDim.x + threadIdx.x;  // 8*KS*32 threads
  const int KS   = K >> 5;
  const int lane = tid & 31;
  const int ks   = (tid >> 5) % KS;
  const int nt   = (tid >> 5) / KS;
  const int n    = nt * 16 + (lane & 15);
  const int kh   = ks * 32 + 8 * (lane >> 4);
  v16h r;
#pragma unroll
  for (int h = 0; h < 16; ++h) {
    const int k = kh + h + (h >= 8 ? 8 : 0);
    r[h] = (_Float16)W[(size_t)n * K + k];
  }
  ((v16h*)out)[tid] = r;
}

// ------------------------- WMMA A-operand helpers --------------------------
__device__ __forceinline__ v16h pack16(float4 a, float4 b, float4 c, float4 d) {
  v16h r;
  r[0]  = (_Float16)a.x; r[1]  = (_Float16)a.y; r[2]  = (_Float16)a.z; r[3]  = (_Float16)a.w;
  r[4]  = (_Float16)b.x; r[5]  = (_Float16)b.y; r[6]  = (_Float16)b.z; r[7]  = (_Float16)b.w;
  r[8]  = (_Float16)c.x; r[9]  = (_Float16)c.y; r[10] = (_Float16)c.z; r[11] = (_Float16)c.w;
  r[12] = (_Float16)d.x; r[13] = (_Float16)d.y; r[14] = (_Float16)d.z; r[15] = (_Float16)d.w;
  return r;
}

// float4 from logical concat([A0|A1]) row m, cols k..k+3 (4-aligned segments
// never straddle the 128 boundary)
__device__ __forceinline__ float4 ld4cc(const float* __restrict__ A0,
                                        const float* __restrict__ A1,
                                        int m, int k) {
  const float* p = (k < CDIM) ? (A0 + (size_t)m * CDIM + k)
                              : (A1 + (size_t)m * CDIM + (k - CDIM));
  return *(const float4*)p;
}

// fp32 H with BN scale/shift + ReLU applied, as float4
__device__ __forceinline__ float4 ld4bn(const float* __restrict__ H,
                                        const float* __restrict__ ss,
                                        int m, int k) {
  float4 h = *(const float4*)(H + (size_t)m * CDIM + k);
  const float4 sc = *(const float4*)(ss + k);
  const float4 sh = *(const float4*)(ss + CDIM + k);
  h.x = fmaxf(fmaf(h.x, sc.x, sh.x), 0.0f);
  h.y = fmaxf(fmaf(h.y, sc.y, sh.y), 0.0f);
  h.z = fmaxf(fmaf(h.z, sc.z, sh.z), 0.0f);
  h.w = fmaxf(fmaf(h.w, sc.w, sh.w), 0.0f);
  return h;
}

// ---------------- GEMM 1: H = concat(A0,A1) @ W^T, + BN stats --------------
// K = 256. Wave strip: 16 rows x 64 cols (4 n-tiles, 4 accumulators), so the
// converted A fragment feeds 4 WMMAs. grid (mtiles/4, 2), block 128.
__global__ void gemm256_f16_stats_kernel(const float* __restrict__ A0,
                                         const float* __restrict__ A1,
                                         const v16h* __restrict__ WF,  // packed
                                         float* __restrict__ H,
                                         float* __restrict__ stats) {
  const int lane = threadIdx.x & 31;
  const int wave = threadIdx.x >> 5;
  const int mt = blockIdx.x * 4 + wave;
  const int ntBase = blockIdx.y * 4;
  const int m  = mt * 16 + (lane & 15);
  const int kh = 8 * (lane >> 4);
  const int KS = 8;                       // 256/32

  v8f acc[4] = {{}, {}, {}, {}};
#pragma unroll
  for (int ks = 0; ks < KS; ++ks) {
    const int k = ks * 32 + kh;
    v16h av = pack16(ld4cc(A0, A1, m, k),      ld4cc(A0, A1, m, k + 4),
                     ld4cc(A0, A1, m, k + 16), ld4cc(A0, A1, m, k + 20));
#pragma unroll
    for (int t = 0; t < 4; ++t) {
      const v16h bv = WF[(size_t)((ntBase + t) * KS + ks) * 32 + lane];
      acc[t] = __builtin_amdgcn_wmma_f32_16x16x32_f16(false, av, false, bv,
                                                      (short)0, acc[t], false, false);
    }
  }
  const int m0 = mt * 16 + 8 * (lane >> 4);
#pragma unroll
  for (int t = 0; t < 4; ++t) {
    const int n = (ntBase + t) * 16 + (lane & 15);
    float s = 0.0f, sq = 0.0f;
#pragma unroll
    for (int r = 0; r < 8; ++r) {
      const float v = acc[t][r];
      H[(size_t)(m0 + r) * CDIM + n] = v;
      s += v; sq += v * v;
    }
    atomicAdd(&stats[n], s);
    atomicAdd(&stats[CDIM + n], sq);
  }
}

// ---------------- BN finalize: scale/shift per channel ---------------------
__global__ void bn_finalize_kernel(const float* __restrict__ stats,
                                   const float* __restrict__ g,
                                   const float* __restrict__ b,
                                   float* __restrict__ ss, float invN) {
  const int c = threadIdx.x;            // 128 threads
  const float mu  = stats[c] * invN;
  const float var = stats[CDIM + c] * invN - mu * mu;
  const float sc  = g[c] * rsqrtf(var + 1e-5f);
  ss[c] = sc;
  ss[CDIM + c] = b[c] - mu * sc;
}

// ------- GEMM 2: out = relu(H*scale+shift) @ W^T + bias, K = 128 -----------
__global__ void gemm128_bn_kernel(const float* __restrict__ H,
                                  const float* __restrict__ ss,
                                  const v16h* __restrict__ WF,   // packed
                                  const float* __restrict__ bias,
                                  float* __restrict__ out) {
  const int lane = threadIdx.x & 31;
  const int wave = threadIdx.x >> 5;
  const int mt = blockIdx.x * 4 + wave;
  const int ntBase = blockIdx.y * 4;
  const int m  = mt * 16 + (lane & 15);
  const int kh = 8 * (lane >> 4);
  const int KS = 4;                       // 128/32

  v8f acc[4] = {{}, {}, {}, {}};
#pragma unroll
  for (int ks = 0; ks < KS; ++ks) {
    const int k = ks * 32 + kh;
    v16h av = pack16(ld4bn(H, ss, m, k),      ld4bn(H, ss, m, k + 4),
                     ld4bn(H, ss, m, k + 16), ld4bn(H, ss, m, k + 20));
#pragma unroll
    for (int t = 0; t < 4; ++t) {
      const v16h bv = WF[(size_t)((ntBase + t) * KS + ks) * 32 + lane];
      acc[t] = __builtin_amdgcn_wmma_f32_16x16x32_f16(false, av, false, bv,
                                                      (short)0, acc[t], false, false);
    }
  }
  const int m0 = mt * 16 + 8 * (lane >> 4);
#pragma unroll
  for (int t = 0; t < 4; ++t) {
    const int n = (ntBase + t) * 16 + (lane & 15);
    const float bn = bias[n];
#pragma unroll
    for (int r = 0; r < 8; ++r)
      out[(size_t)(m0 + r) * CDIM + n] = acc[t][r] + bn;
  }
}

// ---------------------------------------------------------------------------
extern "C" void kernel_launch(void* const* d_in, const int* in_sizes, int n_in,
                              void* d_out, int out_size, void* d_ws, size_t ws_size,
                              hipStream_t stream) {
  (void)in_sizes; (void)n_in; (void)out_size; (void)ws_size;
  const float* pts_r1 = (const float*)d_in[0];
  const float* pts_r2 = (const float*)d_in[1];
  const float* pts_r4 = (const float*)d_in[2];
  const float* feat0  = (const float*)d_in[3];
  const float* feat1  = (const float*)d_in[4];
  const float* feat2  = (const float*)d_in[5];
  const float* w3a = (const float*)d_in[6];
  const float* g3  = (const float*)d_in[7];
  const float* b3  = (const float*)d_in[8];
  const float* w3b = (const float*)d_in[9];
  const float* bb3 = (const float*)d_in[10];
  const float* w4a = (const float*)d_in[11];
  const float* g4  = (const float*)d_in[12];
  const float* b4  = (const float*)d_in[13];
  const float* w4b = (const float*)d_in[14];
  const float* bb4 = (const float*)d_in[15];

  const int B = 2, N1 = 8192, N2 = 4096, N4 = 2048;
  const int NT1 = B * N2;   // 8192  (fnode-3 rows)
  const int NT2 = B * N1;   // 16384 (fnode-4 rows)

  // workspace layout (256B aligned slabs)
  char* base = (char*)d_ws;
  size_t off = 0;
  auto take = [&](size_t bytes) -> char* {
    char* p = base + off;
    off = (off + bytes + 255) & ~(size_t)255;
    return p;
  };
  int*      idx1   = (int*)     take((size_t)NT1 * 3 * 4);
  float*    wgt1   = (float*)   take((size_t)NT1 * 3 * 4);
  float*    f2i    = (float*)   take((size_t)NT1 * CDIM * 4);
  float*    H1     = (float*)   take((size_t)NT1 * CDIM * 4);
  float*    stats1 = (float*)   take(2 * CDIM * 4);
  float*    ss1    = (float*)   take(2 * CDIM * 4);
  float*    n3     = (float*)   take((size_t)NT1 * CDIM * 4);
  int*      idx2   = (int*)     take((size_t)NT2 * 3 * 4);
  float*    wgt2   = (float*)   take((size_t)NT2 * 3 * 4);
  float*    n3i    = (float*)   take((size_t)NT2 * CDIM * 4);
  float*    H2     = (float*)   take((size_t)NT2 * CDIM * 4);
  float*    stats2 = (float*)   take(2 * CDIM * 4);
  float*    ss2    = (float*)   take(2 * CDIM * 4);
  _Float16* w3aP   = (_Float16*)take((size_t)8 * 8 * 32 * 16 * 2);  // K=256
  _Float16* w3bP   = (_Float16*)take((size_t)8 * 4 * 32 * 16 * 2);  // K=128
  _Float16* w4aP   = (_Float16*)take((size_t)8 * 8 * 32 * 16 * 2);
  _Float16* w4bP   = (_Float16*)take((size_t)8 * 4 * 32 * 16 * 2);

  // weight pre-swizzle (tiny; runs every launch, deterministic)
  packW_kernel<<<8, 256, 0, stream>>>(w3a, w3aP, 256);
  packW_kernel<<<4, 256, 0, stream>>>(w3b, w3bP, 128);
  packW_kernel<<<8, 256, 0, stream>>>(w4a, w4aP, 256);
  packW_kernel<<<4, 256, 0, stream>>>(w4b, w4bP, 128);

  // ---- fnode 3: interpolate r4 -> r2, then fc_block on [feat1 | f2i] ----
  knn3_kernel<<<NT1 / 256, 256, 0, stream>>>(pts_r2, pts_r4, N2, N4, idx1, wgt1);
  interp3_kernel<<<(NT1 * 32) / 256, 256, 0, stream>>>(feat2, idx1, wgt1, f2i, NT1);
  hipMemsetAsync(stats1, 0, 2 * CDIM * 4, stream);
  gemm256_f16_stats_kernel<<<dim3(NT1 / 64, 2), 128, 0, stream>>>(
      feat1, f2i, (const v16h*)w3aP, H1, stats1);
  bn_finalize_kernel<<<1, 128, 0, stream>>>(stats1, g3, b3, ss1, 1.0f / (float)NT1);
  gemm128_bn_kernel<<<dim3(NT1 / 64, 2), 128, 0, stream>>>(
      H1, ss1, (const v16h*)w3bP, bb3, n3);

  // ---- fnode 4: interpolate r2 -> r1, then fc_block on [feat0 | n3i] ----
  knn3_kernel<<<NT2 / 256, 256, 0, stream>>>(pts_r1, pts_r2, N1, N2, idx2, wgt2);
  interp3_kernel<<<(NT2 * 32) / 256, 256, 0, stream>>>(n3, idx2, wgt2, n3i, NT2);
  hipMemsetAsync(stats2, 0, 2 * CDIM * 4, stream);
  gemm256_f16_stats_kernel<<<dim3(NT2 / 64, 2), 128, 0, stream>>>(
      feat0, n3i, (const v16h*)w4aP, H2, stats2);
  bn_finalize_kernel<<<1, 128, 0, stream>>>(stats2, g4, b4, ss2, 1.0f / (float)NT2);
  gemm128_bn_kernel<<<dim3(NT2 / 64, 2), 128, 0, stream>>>(
      H2, ss2, (const v16h*)w4bP, bb4, (float*)d_out);
}
